// VectorQuantizer_47167330844771
// MI455X (gfx1250) — matrix operand order repaired and checked by reading, not compiled, not run on hardware
//
#include <hip/hip_runtime.h>
#include <math.h>

// ---------------------------------------------------------------------------
// VQ-VAE codebook lookup, fused for MI455X (gfx1250, wave32, WMMA).
//   x: [128,2048,64] f32   W: [512,64] f32
//   out = quantized [128*2048*64] f32  ++  loss [1] f32  ++  indices [262144] f32
// ---------------------------------------------------------------------------

typedef __attribute__((ext_vector_type(16))) _Float16 v16h;
typedef __attribute__((ext_vector_type(8)))  float    v8f;

#define DIM   64
#define KCB   512            // codebook entries
#define ROWS  16             // rows per wave tile (WMMA M)
#define WPB   8              // waves per block (256 threads)
#define NROW  (128 * 2048)   // total rows
#define NTILE (NROW / ROWS)  // 16384 wave tiles
#define EPSV  1e-7f

// ---- prep: W(f32) -> Wh(f16) row-major + per-codeword squared norms -------
__global__ __launch_bounds__(256)
void vq_prep(const float* __restrict__ W, _Float16* __restrict__ Wh,
             float* __restrict__ wnorm) {
    int row = blockIdx.x * blockDim.x + threadIdx.x;
    if (row >= KCB) return;
    const float* wr = W + row * DIM;
    _Float16* out = Wh + row * DIM;
    float s = 0.f;
    #pragma unroll
    for (int k = 0; k < DIM; ++k) {
        float v = wr[k];
        s += v * v;
        out[k] = (_Float16)v;
    }
    wnorm[row] = s;
}

// ---- main fused kernel ----------------------------------------------------
__global__ __launch_bounds__(256)
void vq_main(const float* __restrict__ x, const float* __restrict__ W,
             const _Float16* __restrict__ Wh, const float* __restrict__ wnorm,
             float* __restrict__ qout, float* __restrict__ idxout,
             float* __restrict__ lossPart) {
    __shared__ _Float16 lds_xn[WPB][ROWS][DIM];   // 16 KB: normalized rows (f16)
    __shared__ float    lds_min[WPB][ROWS][16];   //  8 KB: per-column mins
    __shared__ int      lds_idx[WPB][ROWS][16];   //  8 KB: per-column argmins
    __shared__ int      lds_sel[WPB][ROWS];       //  selected codeword per row

    const int lane = threadIdx.x & 31;
    const int wave = threadIdx.x >> 5;
    const long tile = (long)blockIdx.x * WPB + wave;   // 0..NTILE-1
    const long row0 = tile * ROWS;

    // ---- phase 1: per-row mean / unbiased std; lane = (row r, half h) ----
    const int r = lane & 15;
    const int h = lane >> 4;
    const float* xrow = x + (row0 + r) * DIM + h * 32;

    float xv[32];
    #pragma unroll
    for (int i = 0; i < 8; ++i) {
        float4 t = ((const float4*)xrow)[i];
        xv[4*i+0] = t.x; xv[4*i+1] = t.y; xv[4*i+2] = t.z; xv[4*i+3] = t.w;
    }
    float s = 0.f, ss = 0.f;
    #pragma unroll
    for (int i = 0; i < 32; ++i) { s += xv[i]; ss += xv[i] * xv[i]; }
    s  += __shfl_xor(s, 16);
    ss += __shfl_xor(ss, 16);
    const float mean    = s * (1.f / 64.f);
    const float var     = (ss - s * mean) * (1.f / 63.f);   // unbiased
    const float std_eps = sqrtf(fmaxf(var, 0.f)) + EPSV;
    const float inv     = 1.f / std_eps;

    {   // normalize -> f16 -> LDS (row-major [16][64])
        _Float16 xh[32];
        #pragma unroll
        for (int i = 0; i < 32; ++i) xh[i] = (_Float16)((xv[i] - mean) * inv);
        float4* dst = (float4*)&lds_xn[wave][r][h * 32];
        #pragma unroll
        for (int i = 0; i < 4; ++i) dst[i] = ((float4*)xh)[i];
    }
    __syncthreads();

    // ---- phase 2: build A (ISA 16-bit A 16x32 layout), sweep codebook ----
    // A chunk: lane m<16 holds K=k0..k0+7 and k0+16..k0+23 with k0=0;
    //          lane m+16 holds k0=8.  Second chunk: +32.
    v16h a0, a1;
    {
        const _Float16* base = &lds_xn[wave][lane & 15][0];
        const int k0 = (lane >> 4) * 8;
        ((float4*)&a0)[0] = *(const float4*)(base + k0);
        ((float4*)&a0)[1] = *(const float4*)(base + k0 + 16);
        ((float4*)&a1)[0] = *(const float4*)(base + k0 + 32);
        ((float4*)&a1)[1] = *(const float4*)(base + k0 + 48);
    }

    float minv[8];
    int   mini[8];
    #pragma unroll
    for (int v = 0; v < 8; ++v) { minv[v] = 3.4e38f; mini[v] = 0; }

    const int col = lane & 15;        // B column (codeword within tile)
    const int kb  = (lane >> 4) * 8;  // B: lanes 0-15 K=0..7,16..23; 16-31 K=8..15,24..31

    for (int t = 0; t < KCB / 16; ++t) {
        const int n = t * 16 + col;
        const _Float16* wp = Wh + (long)n * DIM + kb;  // two 8-half chunks, stride 16
        v16h b0, b1;
        ((float4*)&b0)[0] = *(const float4*)(wp);
        ((float4*)&b0)[1] = *(const float4*)(wp + 16);
        ((float4*)&b1)[0] = *(const float4*)(wp + 32);
        ((float4*)&b1)[1] = *(const float4*)(wp + 48);

        v8f c = {};
        c = __builtin_amdgcn_wmma_f32_16x16x32_f16(false, a0, false, b0,
                                                   (short)0, c, false, false);
        c = __builtin_amdgcn_wmma_f32_16x16x32_f16(false, a1, false, b1,
                                                   (short)0, c, false, false);
        const float wn = wnorm[n];
        #pragma unroll
        for (int v = 0; v < 8; ++v) {   // c[v] = dot(row v+8*h2, codeword n)
            float score = wn - 2.f * c[v];
            if (score < minv[v]) { minv[v] = score; mini[v] = n; }
        }
    }

    // per-lane results -> LDS; C layout: lane L holds rows M=v+8*(L>>4), N=L&15
    #pragma unroll
    for (int v = 0; v < 8; ++v) {
        const int m = v + 8 * (lane >> 4);
        lds_min[wave][m][col] = minv[v];
        lds_idx[wave][m][col] = mini[v];
    }
    __syncthreads();

    if (lane < 16) {   // lane reduces one row over 16 columns, first-min ties
        float bv = lds_min[wave][lane][0];
        int   bi = lds_idx[wave][lane][0];
        #pragma unroll
        for (int c16 = 1; c16 < 16; ++c16) {
            float v2 = lds_min[wave][lane][c16];
            int   i2 = lds_idx[wave][lane][c16];
            if (v2 < bv || (v2 == bv && i2 < bi)) { bv = v2; bi = i2; }
        }
        lds_sel[wave][lane] = bi;
        idxout[row0 + lane] = (float)bi;
    }
    __syncthreads();

    // ---- phase 3: gather f32 codeword, denormalize, loss -----------------
    const int sel = lds_sel[wave][r];
    const float* wq = W + (long)sel * DIM + h * 32;
    float* orow = qout + (row0 + r) * DIM + h * 32;
    float lsum = 0.f;
    #pragma unroll
    for (int i = 0; i < 8; ++i) {
        float4 xt = ((const float4*)xrow)[i];   // cache-hot reload
        float4 wt = ((const float4*)wq)[i];
        float xn0 = (xt.x - mean) * inv, xn1 = (xt.y - mean) * inv;
        float xn2 = (xt.z - mean) * inv, xn3 = (xt.w - mean) * inv;
        float d0 = wt.x - xn0, d1 = wt.y - xn1, d2 = wt.z - xn2, d3 = wt.w - xn3;
        lsum += d0*d0 + d1*d1 + d2*d2 + d3*d3;
        float4 o;
        o.x = wt.x * std_eps + mean;  o.y = wt.y * std_eps + mean;
        o.z = wt.z * std_eps + mean;  o.w = wt.w * std_eps + mean;
        ((float4*)orow)[i] = o;
    }
    // deterministic wave reduction of loss partial
    #pragma unroll
    for (int off = 16; off >= 1; off >>= 1) lsum += __shfl_xor(lsum, off);
    if (lane == 0) lossPart[tile] = lsum;
}

// ---- fixed-order final loss reduction (deterministic) ---------------------
__global__ __launch_bounds__(256)
void vq_loss_reduce(const float* __restrict__ part, int n,
                    float* __restrict__ lossOut) {
    __shared__ float sm[256];
    float s = 0.f;
    for (int i = threadIdx.x; i < n; i += 256) s += part[i];
    sm[threadIdx.x] = s;
    __syncthreads();
    for (int off = 128; off >= 1; off >>= 1) {
        if ((int)threadIdx.x < off) sm[threadIdx.x] += sm[threadIdx.x + off];
        __syncthreads();
    }
    if (threadIdx.x == 0) {
        // q_latent + 0.25*e_latent; both equal mean((q-xn)^2) numerically
        lossOut[0] = sm[0] * (1.25f / (float)(NROW * DIM));
    }
}

extern "C" void kernel_launch(void* const* d_in, const int* in_sizes, int n_in,
                              void* d_out, int out_size, void* d_ws, size_t ws_size,
                              hipStream_t stream) {
    const float* x = (const float*)d_in[0];
    const float* W = (const float*)d_in[1];

    float* out    = (float*)d_out;
    float* qout   = out;                          // 16,777,216 floats
    float* loss   = out + (size_t)NROW * DIM;     // 1 float
    float* idxout = loss + 1;                     // 262,144 floats

    char* ws = (char*)d_ws;
    _Float16* Wh     = (_Float16*)ws;                              // 64 KB
    float*    wnorm  = (float*)(ws + (size_t)KCB * DIM * 2);       //  2 KB
    float*    part   = (float*)(ws + (size_t)KCB * DIM * 2 + KCB * 4); // 64 KB

    vq_prep<<<(KCB + 255) / 256, 256, 0, stream>>>(W, Wh, wnorm);
    vq_main<<<NROW / (ROWS * WPB), 256, 0, stream>>>(x, W, Wh, wnorm,
                                                     qout, idxout, part);
    vq_loss_reduce<<<1, 256, 0, stream>>>(part, NTILE, loss);
}
